// WindowAttention_25271587570167
// MI455X (gfx1250) — matrix operand order repaired and checked by reading, not compiled, not run on hardware
//
#include <hip/hip_runtime.h>

#define DIM    128
#define HEADS  4
#define HD     32
#define WS     7
#define NTOK   49
#define NPAD   64
#define NWIN   64

// LDS row strides (in f16 elements); all rows 16B-aligned, padded vs bank conflicts
#define XS 136
#define QS 40
#define VS 72
#define SS 72
#define OS 136

typedef __attribute__((ext_vector_type(16))) _Float16 v16h;
typedef __attribute__((ext_vector_type(8)))  _Float16 v8h;
typedef __attribute__((ext_vector_type(8)))  float    v8f;

// Load a 16-element f16 fragment as two 16B chunks (chunk1 at +off2 halves).
// A-matrix: pass p = row_base + kb + 8*hsel, off2 = 16  (lane half holds K {0..7,16..23} / {8..15,24..31})
// B-matrix: pass p = col_base + kb + 8*hsel, off2 = 16  (same K layout as A)
__device__ __forceinline__ v16h frag_ld(const _Float16* p, int off2) {
  v8h c0 = *(const v8h*)p;
  v8h c1 = *(const v8h*)(p + off2);
  v16h r;
#pragma unroll
  for (int i = 0; i < 8; ++i) { r[i] = c0[i]; r[8 + i] = c1[i]; }
  return r;
}

__global__ void cvt_weights(const float* __restrict__ qkv_w,
                            const float* __restrict__ proj_w,
                            _Float16* __restrict__ qkv_wh,
                            _Float16* __restrict__ proj_wh) {
  int i = blockIdx.x * blockDim.x + threadIdx.x;
  if (i < 3 * DIM * DIM) qkv_wh[i] = (_Float16)qkv_w[i];
  if (i < DIM * DIM)     proj_wh[i] = (_Float16)proj_w[i];
}

__global__ __launch_bounds__(256)
void win_attn(const float* __restrict__ x,
              const float* __restrict__ mask,
              const _Float16* __restrict__ qkv_wh,
              const float* __restrict__ qkv_b,
              const _Float16* __restrict__ proj_wh,
              const float* __restrict__ proj_b,
              const float* __restrict__ bias_table,
              float* __restrict__ out)
{
  __shared__ __align__(16) _Float16 Xls[NPAD * XS];  // x tile, f16 (reused all heads)
  __shared__ __align__(16) _Float16 Qls[NPAD * QS];  // per-head Q [64x32]
  __shared__ __align__(16) _Float16 Kls[NPAD * QS];  // per-head K [64x32]
  __shared__ __align__(16) _Float16 Vts[HD * VS];    // per-head V^T [32x64]
  __shared__ __align__(16) _Float16 Sp [NPAD * SS];  // scores S, then probs P (in place)
  __shared__ __align__(16) _Float16 Ol [NPAD * OS];  // attention output [64x128]
  __shared__ float rs[NPAD];                          // per-row 1/sum for softmax

  const int b    = blockIdx.x;
  const int tid  = threadIdx.x;
  const int wave = tid >> 5;
  const int lane = tid & 31;
  const int nn   = lane & 15;   // A row / B column / C column within tile
  const int hsel = lane >> 4;

  // ---- Phase 0: stage x window into LDS as f16, zero-pad rows 49..63 ----
  const float* xp = x + (size_t)b * (NTOK * DIM);
  for (int idx = tid; idx < NTOK * DIM; idx += 256) {
    int r = idx >> 7, c = idx & 127;
    Xls[r * XS + c] = (_Float16)xp[idx];
  }
  for (int idx = tid; idx < (NPAD - NTOK) * DIM; idx += 256) {
    int r = NTOK + (idx >> 7), c = idx & 127;
    Xls[r * XS + c] = (_Float16)0.f;
  }
  __syncthreads();

  for (int h = 0; h < HEADS; ++h) {
    // ---- QKV projection for this head: 4 row-tiles x 6 col-tiles, K=128 ----
    for (int t = wave; t < 24; t += 8) {
      const int r    = t & 3;
      const int gc   = t >> 2;          // 0..5
      const int sel  = gc >> 1;         // 0=Q 1=K 2=V
      const int cl   = (gc & 1) << 4;   // 0 or 16
      const int wrow = sel * DIM + h * HD + cl;  // qkv_w row base (B^T rows)
      v8f acc = {};
#pragma unroll
      for (int kt = 0; kt < 4; ++kt) {
        const int kb = kt << 5;
        v16h a  = frag_ld(&Xls[(r * 16 + nn) * XS + kb + 8 * hsel], 16);
        v16h bf = frag_ld(&qkv_wh[(size_t)(wrow + nn) * DIM + kb + 8 * hsel], 16);
        acc = __builtin_amdgcn_wmma_f32_16x16x32_f16(false, a, false, bf,
                                                     (short)0, acc, false, false);
      }
      const float bias = qkv_b[wrow + nn];
      if (sel < 2) {
        _Float16* dst = (sel == 0) ? Qls : Kls;
#pragma unroll
        for (int i = 0; i < 8; ++i) {
          int mm = r * 16 + i + 8 * hsel;
          dst[mm * QS + cl + nn] = (_Float16)(acc[i] + bias);
        }
      } else {  // V stored transposed: Vts[dim][token]
#pragma unroll
        for (int i = 0; i < 8; ++i) {
          int mm = r * 16 + i + 8 * hsel;
          Vts[(cl + nn) * VS + mm] = (_Float16)(acc[i] + bias);
        }
      }
    }
    __syncthreads();

    // ---- S = Q K^T : 16 tiles, one 16x16x32 WMMA each (K = head_dim = 32) ----
    for (int t = wave; t < 16; t += 8) {
      const int r = t & 3, c = t >> 2;
      v16h a  = frag_ld(&Qls[(r * 16 + nn) * QS + 8 * hsel], 16);
      v16h bf = frag_ld(&Kls[(c * 16 + nn) * QS + 8 * hsel], 16);
      v8f acc = {};
      acc = __builtin_amdgcn_wmma_f32_16x16x32_f16(false, a, false, bf,
                                                   (short)0, acc, false, false);
#pragma unroll
      for (int i = 0; i < 8; ++i) {
        int mm = r * 16 + i + 8 * hsel;
        Sp[mm * SS + c * 16 + nn] = (_Float16)acc[i];
      }
    }
    __syncthreads();

    // ---- softmax(scale*S + rel_pos_bias + mask), one row per thread ----
    if (tid < NPAD) rs[tid] = 0.f;     // rows 49..63 -> zero out O rows
    if (tid < NTOK) {
      const int i  = tid;
      const int iy = i / WS, ix = i - iy * WS;
      const float scale = 0.17677669529663687f;  // 1/sqrt(32)
      const float* mrow = mask + ((size_t)(b & (NWIN - 1)) * NTOK + i) * NTOK;
      _Float16* srow = &Sp[i * SS];
      float mx = -1e30f;
      int jy = 0, jx = 0;
      for (int j = 0; j < NTOK; ++j) {
        int rel = (iy - jy + WS - 1) * (2 * WS - 1) + (ix - jx + WS - 1);
        float l = (float)srow[j] * scale + bias_table[rel * HEADS + h] + mrow[j];
        srow[j] = (_Float16)l;
        mx = fmaxf(mx, l);
        if (++jx == WS) { jx = 0; ++jy; }
      }
      float sum = 0.f;
      for (int j = 0; j < NTOK; ++j) {
        float e = __expf((float)srow[j] - mx);
        sum += e;
        srow[j] = (_Float16)e;         // P (unnormalized) in place over S
      }
      for (int j = NTOK; j < NPAD; ++j) srow[j] = (_Float16)0.f;  // kill pad cols
      rs[i] = 1.0f / sum;              // normalization folded into PV store
    }
    __syncthreads();

    // ---- O_h = P V : 8 tiles (one per wave), K = 64 tokens ----
    {
      const int r = wave & 3, c = wave >> 2;
      v8f acc = {};
#pragma unroll
      for (int kt = 0; kt < 2; ++kt) {
        const int kb = kt << 5;
        v16h a  = frag_ld(&Sp[(r * 16 + nn) * SS + kb + 8 * hsel], 16);
        v16h bf = frag_ld(&Vts[(c * 16 + nn) * VS + kb + 8 * hsel], 16);
        acc = __builtin_amdgcn_wmma_f32_16x16x32_f16(false, a, false, bf,
                                                     (short)0, acc, false, false);
      }
#pragma unroll
      for (int i = 0; i < 8; ++i) {
        int mm = r * 16 + i + 8 * hsel;
        Ol[mm * OS + h * HD + c * 16 + nn] = (_Float16)(acc[i] * rs[mm]);
      }
    }
    __syncthreads();
  }

  // ---- output projection: Y = O * proj_w^T + proj_b, 32 tiles, K=128 ----
  for (int t = wave; t < 32; t += 8) {
    const int r = t & 3, c = t >> 2;
    v8f acc = {};
#pragma unroll
    for (int kt = 0; kt < 4; ++kt) {
      const int kb = kt << 5;
      v16h a  = frag_ld(&Ol[(r * 16 + nn) * OS + kb + 8 * hsel], 16);
      v16h bf = frag_ld(&proj_wh[(size_t)(c * 16 + nn) * DIM + kb + 8 * hsel], 16);
      acc = __builtin_amdgcn_wmma_f32_16x16x32_f16(false, a, false, bf,
                                                   (short)0, acc, false, false);
    }
    const float pb = proj_b[c * 16 + nn];
#pragma unroll
    for (int i = 0; i < 8; ++i) {
      int mm = r * 16 + i + 8 * hsel;
      if (mm < NTOK)
        out[((size_t)b * NTOK + mm) * DIM + c * 16 + nn] = acc[i] + pb;
    }
  }
}

extern "C" void kernel_launch(void* const* d_in, const int* in_sizes, int n_in,
                              void* d_out, int out_size, void* d_ws, size_t ws_size,
                              hipStream_t stream) {
  const float* x        = (const float*)d_in[0];
  const float* mask     = (const float*)d_in[1];
  const float* qkv_w    = (const float*)d_in[2];
  const float* qkv_b    = (const float*)d_in[3];
  const float* proj_w   = (const float*)d_in[4];
  const float* proj_b   = (const float*)d_in[5];
  const float* bias_tab = (const float*)d_in[6];
  float* out = (float*)d_out;

  _Float16* qkv_wh  = (_Float16*)d_ws;
  _Float16* proj_wh = (_Float16*)((char*)d_ws + (size_t)3 * DIM * DIM * sizeof(_Float16));

  cvt_weights<<<(3 * DIM * DIM + 255) / 256, 256, 0, stream>>>(qkv_w, proj_w, qkv_wh, proj_wh);
  win_attn<<<4096, 256, 0, stream>>>(x, mask, qkv_wh, qkv_b, proj_wh, proj_b, bias_tab, out);
}
